// SequenceMoE_46651934769872
// MI455X (gfx1250) — matrix operand order repaired and checked by reading, not compiled, not run on hardware
//
#include <hip/hip_runtime.h>
#include <stdint.h>
#include <math.h>

#define Bb 8
#define Tt 1024
#define Cc 1024
#define Ee 8
#define Kk 2
#define Hh 4096
#define KH 256          // h-chunk width staged in LDS

typedef __bf16 bf16_t;
typedef __attribute__((ext_vector_type(8)))  bf16_t v8bf;
typedef __attribute__((ext_vector_type(16))) bf16_t v16bf;
typedef __attribute__((ext_vector_type(8)))  float  v8f;

__device__ __forceinline__ uint16_t f2bf(float f) {
    uint32_t u = __float_as_uint(f);
    u += 0x7FFFu + ((u >> 16) & 1u);        // round-to-nearest-even
    return (uint16_t)(u >> 16);
}

// Branch-free tanh-approx GELU: tanh(u) = 1 - 2/(1 + e^{2u}),
// e^{2u} = exp2(u * 2*log2(e)) via raw v_exp_f32; saturates correctly at +/-inf.
__device__ __forceinline__ float gelu_fast(float v) {
    float u = 0.7978845608028654f * (v + 0.044715f * v * v * v);
    float e = __builtin_amdgcn_exp2f(u * 2.885390081777927f);  // e^{2u}
    float t = 1.0f - 2.0f * __builtin_amdgcn_rcpf(1.0f + e);   // tanh(u)
    return 0.5f * v * (1.0f + t);
}

// A fragment (16x32 bf16): lane m=lane&15; K chunks of 8 at k0 and k0+16.
__device__ __forceinline__ v16bf load_a16(const uint16_t* __restrict__ rowPtr, int k0) {
    v8bf lo = *reinterpret_cast<const v8bf*>(rowPtr + k0);
    v8bf hi = *reinterpret_cast<const v8bf*>(rowPtr + k0 + 16);
    return __builtin_shufflevector(lo, hi, 0,1,2,3,4,5,6,7,8,9,10,11,12,13,14,15);
}
// B fragment (32x16 bf16): lane n=lane&15; K chunks of 8 at k0 and k0+16.
__device__ __forceinline__ v16bf load_b16(const uint16_t* __restrict__ colPtr, int k0) {
    v8bf lo = *reinterpret_cast<const v8bf*>(colPtr + k0);
    v8bf hi = *reinterpret_cast<const v8bf*>(colPtr + k0 + 16);
    return __builtin_shufflevector(lo, hi, 0,1,2,3,4,5,6,7,8,9,10,11,12,13,14,15);
}

// ---------------- small kernels ----------------

__global__ __launch_bounds__(256) void seqrep_kernel(const float* __restrict__ x,
                                                     float* __restrict__ seq_rep) {
    int idx = blockIdx.x * 256 + threadIdx.x;          // B*C = 8192
    if (idx >= Bb * Cc) return;
    int b = idx / Cc, c = idx % Cc;
    const float* p = x + (size_t)b * Tt * Cc + c;
    float s = 0.f;
    for (int t = 0; t < Tt; ++t) s += p[(size_t)t * Cc];
    seq_rep[idx] = s * (1.0f / Tt);
}

__global__ __launch_bounds__(64) void router_kernel(const float* __restrict__ seq_rep,
                                                    const float* __restrict__ rw,
                                                    int* __restrict__ sel,
                                                    float* __restrict__ wt,
                                                    float* __restrict__ loss_out) {
    __shared__ float logits[Bb * Ee];
    int tid = threadIdx.x;
    if (tid < Bb * Ee) {
        int b = tid >> 3, e = tid & 7;
        float s = 0.f;
        for (int i = 0; i < Cc; ++i) s += seq_rep[b * Cc + i] * rw[e * Cc + i];
        logits[tid] = s;
    }
    __syncthreads();
    if (tid == 0) {
        float imp[Ee], loadc[Ee];
        for (int e = 0; e < Ee; ++e) { imp[e] = 0.f; loadc[e] = 0.f; }
        for (int b = 0; b < Bb; ++b) {
            float mx = -1e30f;
            for (int e = 0; e < Ee; ++e) mx = fmaxf(mx, logits[b * Ee + e]);
            float p[Ee], s = 0.f;
            for (int e = 0; e < Ee; ++e) { p[e] = expf(logits[b * Ee + e] - mx); s += p[e]; }
            for (int e = 0; e < Ee; ++e) { p[e] /= s; imp[e] += p[e]; }
            int i0 = 0;
            for (int e = 1; e < Ee; ++e) if (p[e] > p[i0]) i0 = e;
            int i1 = (i0 == 0) ? 1 : 0;
            for (int e = 0; e < Ee; ++e) if (e != i0 && p[e] > p[i1]) i1 = e;
            float sw = p[i0] + p[i1];
            sel[b * Kk + 0] = i0;  sel[b * Kk + 1] = i1;
            wt[b * Kk + 0] = p[i0] / sw;  wt[b * Kk + 1] = p[i1] / sw;
            loadc[i0] += 1.f;  loadc[i1] += 1.f;
        }
        float L = 0.f;
        for (int e = 0; e < Ee; ++e) L += (imp[e] / (float)Bb) * (loadc[e] / (float)Bb);
        loss_out[0] = (float)Ee * L;
    }
}

__global__ __launch_bounds__(256) void cvt_x_kernel(const float* __restrict__ x,
                                                    uint16_t* __restrict__ xb) {
    size_t i = (size_t)blockIdx.x * 256 + threadIdx.x;   // B*T*C = 8M
    xb[i] = f2bf(x[i]);
}

// w_fc [E][C][H] -> wfcT [E][H][C]  (bf16)
__global__ __launch_bounds__(256) void trans_fc_kernel(const float* __restrict__ w,
                                                       uint16_t* __restrict__ o) {
    size_t i = (size_t)blockIdx.x * 256 + threadIdx.x;   // E*H*C = 32M
    int c = (int)(i % Cc);
    size_t r = i / Cc;
    int h = (int)(r % Hh);
    int e = (int)(r / Hh);
    o[i] = f2bf(w[((size_t)e * Cc + c) * Hh + h]);
}

// w_proj [E][H][C] -> wprojT [E][C][H]  (bf16)
__global__ __launch_bounds__(256) void trans_pj_kernel(const float* __restrict__ w,
                                                       uint16_t* __restrict__ o) {
    size_t i = (size_t)blockIdx.x * 256 + threadIdx.x;   // E*C*H = 32M
    int h = (int)(i % Hh);
    size_t r = i / Hh;
    int c = (int)(r % Cc);
    int e = (int)(r / Cc);
    o[i] = f2bf(w[((size_t)e * Hh + h) * Cc + c]);
}

// ---------------- fused MoE kernel ----------------
// grid: B * (T/32) blocks, 256 threads (8 waves).
// Block computes out[b][row0:row0+32][0:1024], summed over the 2 selected experts.
// Per 256-wide H chunk: all 8 waves build h = w*gelu(x.w_fc) (2x2 tiles each, A&B
// fragments reused 2x), stage bf16 in LDS, then GEMM2 with 2 row-tiles x 8
// col-tiles per wave (each B fragment reused by both row-tiles).
__global__ __launch_bounds__(256) void moe_fused_kernel(
    const uint16_t* __restrict__ xb,      // [B][T][C] bf16
    const uint16_t* __restrict__ wfcT,    // [E][H][C] bf16
    const uint16_t* __restrict__ wprojT,  // [E][C][H] bf16
    const int*      __restrict__ sel,     // [B][K]
    const float*    __restrict__ wt,      // [B][K] (normalized)
    float*          __restrict__ out)     // [B][T][C]
{
    __shared__ __align__(16) uint16_t hs[32 * KH];   // 16 KB bf16 staging

    const int tid  = threadIdx.x;
    const int lane = tid & 31;
    const int wave = tid >> 5;
    const int b    = blockIdx.x >> 5;          // T/32 = 32 row blocks
    const int row0 = (blockIdx.x & 31) * 32;

    const int n16   = lane & 15;
    const int hi16  = lane >> 4;               // 0 or 1
    const int aksel = hi16 * 8;                // A-fragment K select
    const int bksel = hi16 * 8;                // B-fragment K select

    // acc2[i][t]: row-tile i (0..1), col-tile t (0..7); cols = wave*128 + t*16
    v8f acc2[2][8];
    #pragma unroll
    for (int i = 0; i < 2; ++i)
        #pragma unroll
        for (int t = 0; t < 8; ++t)
            #pragma unroll
            for (int j = 0; j < 8; ++j) acc2[i][t][j] = 0.0f;

    const uint16_t* xrow0 = xb + (size_t)(b * Tt + row0 + n16) * Cc;       // row-tile 0
    const uint16_t* xrow1 = xrow0 + (size_t)16 * Cc;                       // row-tile 1
    const uint16_t* hrow0 = &hs[(n16) * KH];
    const uint16_t* hrow1 = &hs[(16 + n16) * KH];

    for (int k = 0; k < Kk; ++k) {
        const int   e = sel[b * Kk + k];
        const float w = wt[b * Kk + k];
        const uint16_t* bfc = wfcT   + (size_t)e * Hh * Cc;
        const uint16_t* bpj = wprojT + (size_t)e * Cc * Hh
                                     + (size_t)(wave * 128 + n16) * Hh;

        for (int hc = 0; hc < Hh; hc += KH) {
            // ---- GEMM1: 2x2 tiles of h[32 x 256]; wave owns h-cols wave*32..+32 ----
            v8f acc1[2][2];
            #pragma unroll
            for (int i = 0; i < 2; ++i)
                #pragma unroll
                for (int j2 = 0; j2 < 2; ++j2)
                    #pragma unroll
                    for (int j = 0; j < 8; ++j) acc1[i][j2][j] = 0.0f;

            const uint16_t* bcol0 = bfc + (size_t)(hc + wave * 32 + n16) * Cc;
            const uint16_t* bcol1 = bcol0 + (size_t)16 * Cc;
            #pragma unroll 2
            for (int kc = 0; kc < Cc; kc += 32) {
                v16bf a0 = load_a16(xrow0, kc + aksel);
                v16bf a1 = load_a16(xrow1, kc + aksel);
                v16bf b0 = load_b16(bcol0, kc + bksel);
                v16bf b1 = load_b16(bcol1, kc + bksel);
                acc1[0][0] = __builtin_amdgcn_wmma_f32_16x16x32_bf16(
                                 false, a0, false, b0, (short)0, acc1[0][0], false, false);
                acc1[0][1] = __builtin_amdgcn_wmma_f32_16x16x32_bf16(
                                 false, a0, false, b1, (short)0, acc1[0][1], false, false);
                acc1[1][0] = __builtin_amdgcn_wmma_f32_16x16x32_bf16(
                                 false, a1, false, b0, (short)0, acc1[1][0], false, false);
                acc1[1][1] = __builtin_amdgcn_wmma_f32_16x16x32_bf16(
                                 false, a1, false, b1, (short)0, acc1[1][1], false, false);
            }
            // GELU (branch-free), fold routing weight, stage as bf16 in LDS
            #pragma unroll
            for (int i = 0; i < 2; ++i)
                #pragma unroll
                for (int j2 = 0; j2 < 2; ++j2)
                    #pragma unroll
                    for (int j = 0; j < 8; ++j) {
                        float g = gelu_fast(acc1[i][j2][j]) * w;
                        int m = i * 16 + j + (hi16 << 3);       // C/D layout row
                        int col = wave * 32 + j2 * 16 + n16;
                        hs[m * KH + col] = f2bf(g);
                    }
            __syncthreads();

            // ---- GEMM2: out_tile += h_chunk(32x256) * w_proj(256 x 1024) ----
            #pragma unroll 2
            for (int kk = 0; kk < KH; kk += 32) {
                v16bf a0 = load_a16(hrow0, kk + aksel);
                v16bf a1 = load_a16(hrow1, kk + aksel);
                #pragma unroll
                for (int t = 0; t < 8; ++t) {
                    const uint16_t* pcol = bpj + (size_t)t * 16 * Hh;
                    v16bf b2 = load_b16(pcol, hc + kk + bksel);
                    acc2[0][t] = __builtin_amdgcn_wmma_f32_16x16x32_bf16(
                                     false, a0, false, b2, (short)0, acc2[0][t], false, false);
                    acc2[1][t] = __builtin_amdgcn_wmma_f32_16x16x32_bf16(
                                     false, a1, false, b2, (short)0, acc2[1][t], false, false);
                }
            }
            __syncthreads();   // hs reused next chunk
        }
    }

    // Epilogue: write out tile (already weighted & summed over experts)
    float* outp = out + (size_t)b * Tt * Cc;
    #pragma unroll
    for (int i = 0; i < 2; ++i)
        #pragma unroll
        for (int t = 0; t < 8; ++t) {
            int col = wave * 128 + t * 16 + n16;
            #pragma unroll
            for (int j = 0; j < 8; ++j) {
                int row = row0 + i * 16 + j + (hi16 << 3);
                outp[(size_t)row * Cc + col] = acc2[i][t][j];
            }
        }
}

// ---------------- launch ----------------

extern "C" void kernel_launch(void* const* d_in, const int* in_sizes, int n_in,
                              void* d_out, int out_size, void* d_ws, size_t ws_size,
                              hipStream_t stream) {
    const float* x        = (const float*)d_in[0];   // [B,T,C]
    const float* router_w = (const float*)d_in[1];   // [E,C]
    const float* w_fc     = (const float*)d_in[2];   // [E,C,H]
    const float* w_proj   = (const float*)d_in[3];   // [E,H,C]
    float* out = (float*)d_out;                      // [B,T,C] + loss

    char* ws = (char*)d_ws;
    float*    seq_rep = (float*)(ws);                                  // 32 KB
    int*      sel     = (int*)(ws + 32768);
    float*    wt      = (float*)(ws + 32832);
    uint16_t* xb      = (uint16_t*)(ws + 65536);                       // 16 MB
    uint16_t* wfcT    = (uint16_t*)(ws + 65536 + (size_t)16 * 1024 * 1024);  // 64 MB
    uint16_t* wprojT  = wfcT + (size_t)Ee * Hh * Cc;                   // 64 MB

    seqrep_kernel<<<(Bb * Cc + 255) / 256, 256, 0, stream>>>(x, seq_rep);
    router_kernel<<<1, 64, 0, stream>>>(seq_rep, router_w, sel, wt,
                                        out + (size_t)Bb * Tt * Cc);
    cvt_x_kernel<<<(Bb * Tt * Cc) / 256, 256, 0, stream>>>(x, xb);
    trans_fc_kernel<<<((size_t)Ee * Hh * Cc) / 256, 256, 0, stream>>>(w_fc, wfcT);
    trans_pj_kernel<<<((size_t)Ee * Cc * Hh) / 256, 256, 0, stream>>>(w_proj, wprojT);
    moe_fused_kernel<<<Bb * (Tt / 32), 256, 0, stream>>>(xb, wfcT, wprojT, sel, wt, out);
}
